// MultiheadAttention_11991548690802
// MI455X (gfx1250) — matrix operand order repaired and checked by reading, not compiled, not run on hardware
//
#include <hip/hip_runtime.h>

#ifndef __has_builtin
#define __has_builtin(x) 0
#endif

#if __has_builtin(__builtin_amdgcn_global_load_async_to_lds_b128) && \
    __has_builtin(__builtin_amdgcn_s_wait_asynccnt)
#define USE_ASYNC_LDS 1
#else
#define USE_ASYNC_LDS 0
#endif

#define SEQ    1024
#define BATCH  16
#define DMODEL 512
#define HDIM   64
#define NHEAD  8
#define FFDIM  2048
#define MROWS  (SEQ * BATCH)   // 16384

typedef __attribute__((ext_vector_type(16))) __bf16 v16bf;
typedef __attribute__((ext_vector_type(8)))  float  v8f;
typedef __attribute__((ext_vector_type(4)))  int    i32x4;
typedef unsigned short u16;
typedef unsigned int   u32;

// async-copy pointer casts (signature per hipcc diagnostic: v4i* in AS1 / AS3)
#define ASYNC_G(p) ((__attribute__((address_space(1))) i32x4*)(p))
#define ASYNC_L(p) ((__attribute__((address_space(3))) i32x4*)(p))

struct alignas(16) Q16 { u32 a, b, c, d; };

union FragA {
    Q16   q[2];
    u16   u[16];
    v16bf v;
};

__device__ __forceinline__ u16 f2bf(float f) {
    union { float f; u32 u; } v; v.f = f;
    u32 r = v.u + 0x7FFFu + ((v.u >> 16) & 1u);   // round-to-nearest-even
    return (u16)(r >> 16);
}

__device__ __forceinline__ v8f wmma_bf16(v16bf a, v16bf b, v8f c) {
    return __builtin_amdgcn_wmma_f32_16x16x32_bf16(
        /*neg_a=*/false, a, /*neg_b=*/false, b,
        /*c_mod=*/(short)0, c, /*reuse_a=*/false, /*reuse_b=*/false);
}

// ---------------------------------------------------------------------------
// sequence [S,B,D] f32 -> Xbf [B*S, D] bf16
// ---------------------------------------------------------------------------
__global__ void pack_seq_kernel(const float* __restrict__ seq, u16* __restrict__ xb) {
    long idx = (long)blockIdx.x * blockDim.x + threadIdx.x;
    if (idx >= (long)SEQ * BATCH * DMODEL) return;
    int d = (int)(idx % DMODEL);
    int b = (int)((idx / DMODEL) % BATCH);
    int s = (int)(idx / ((long)DMODEL * BATCH));
    xb[((long)b * SEQ + s) * DMODEL + d] = f2bf(seq[idx]);
}

// Wq/Wk/Wv [H,D,HD] f32 -> TRANSPOSED bf16 [H*HD][D]  (Bt[n][k], n=h*HD+e, k=d)
__global__ void repack_qkv_kernel(const float* __restrict__ w, u16* __restrict__ o) {
    int idx = blockIdx.x * blockDim.x + threadIdx.x;
    if (idx >= NHEAD * DMODEL * HDIM) return;
    int e = idx % HDIM;
    int d = (idx / HDIM) % DMODEL;
    int h = idx / (HDIM * DMODEL);
    o[((long)(h * HDIM + e)) * DMODEL + d] = f2bf(w[idx]);
}

// in [K][N] f32 -> out [N][K] bf16 via 32x32 LDS tile (K,N multiples of 32)
__global__ void transpose_cvt_kernel(const float* __restrict__ in, u16* __restrict__ out,
                                     int K, int N) {
    __shared__ float tile[32][33];
    int n0 = blockIdx.x * 32, k0 = blockIdx.y * 32;
    int tx = threadIdx.x, ty = threadIdx.y;   // 32 x 8
    #pragma unroll
    for (int i = 0; i < 32; i += 8)
        tile[ty + i][tx] = in[(long)(k0 + ty + i) * N + n0 + tx];
    __syncthreads();
    #pragma unroll
    for (int i = 0; i < 32; i += 8)
        out[(long)(n0 + ty + i) * K + k0 + tx] = f2bf(tile[tx][ty + i]);
}

// ---------------------------------------------------------------------------
// GEMM: C[M,N] = act( (A_bf16[M,K] x B + bias)*scale + res ), B given
// PRE-TRANSPOSED as Bt[N][K] bf16. Block: 256 threads (8 waves), tile 128x128,
// K-step 32 through LDS; all LDS traffic is 16B vectors.
// RES_MODE: 0 none, 1 res=f32[M,N], 2 res=sequence layout [S,B,D]
// OUT_SEQ : f32 output in [S,B,D] layout (final y)
// OUTB_VT : bf16 output written as Vt[b][h][e][s] (for attention V)
// ---------------------------------------------------------------------------
template<int RES_MODE, bool RELU, bool OUT_SEQ, bool OUTB_VT>
__global__ __launch_bounds__(256) void gemm_bf16_kernel(
    const u16* __restrict__ A, const u16* __restrict__ Bt,
    const float* __restrict__ bias, const float* __restrict__ res,
    float* __restrict__ outF, u16* __restrict__ outB,
    int N, int K, float scale)
{
    __shared__ __align__(32) u16 sA[128 * 32];    // [row][k]
    __shared__ __align__(32) u16 sB[128 * 32];    // [n][k]

    const int tid  = threadIdx.x;
    const int wave = tid >> 5;
    const int lane = tid & 31;
    const int hl   = lane >> 4;
    const int ln   = lane & 15;
    const long row0 = (long)blockIdx.y * 128;
    const int  col0 = blockIdx.x * 128;

    v8f acc[8] = {};

    for (int k0 = 0; k0 < K; k0 += 32) {
        // stage A tile: thread -> row tid>>1, 16 elems at (tid&1)*16
        {
            int r  = tid >> 1;
            int kb = (tid & 1) * 16;
            const u16* src = A + (row0 + r) * K + k0 + kb;
#if USE_ASYNC_LDS
            __builtin_amdgcn_global_load_async_to_lds_b128(
                ASYNC_G(src),     ASYNC_L(&sA[r * 32 + kb]),     0, 0);
            __builtin_amdgcn_global_load_async_to_lds_b128(
                ASYNC_G(src + 8), ASYNC_L(&sA[r * 32 + kb + 8]), 0, 0);
#else
            Q16 a0 = *(const Q16*)src;
            Q16 a1 = *(const Q16*)(src + 8);
            *(Q16*)(&sA[r * 32 + kb])     = a0;
            *(Q16*)(&sA[r * 32 + kb + 8]) = a1;
#endif
            if (k0 + 32 < K) __builtin_prefetch(src + 32, 0, 1);
        }
        // stage Bt tile (already [n][k] in global): coalesced copy
        {
            int n  = tid >> 1;
            int kb = (tid & 1) * 16;
            const u16* src = Bt + (long)(col0 + n) * K + k0 + kb;
#if USE_ASYNC_LDS
            __builtin_amdgcn_global_load_async_to_lds_b128(
                ASYNC_G(src),     ASYNC_L(&sB[n * 32 + kb]),     0, 0);
            __builtin_amdgcn_global_load_async_to_lds_b128(
                ASYNC_G(src + 8), ASYNC_L(&sB[n * 32 + kb + 8]), 0, 0);
            __builtin_amdgcn_s_wait_asynccnt(0);
#else
            Q16 b0 = *(const Q16*)src;
            Q16 b1 = *(const Q16*)(src + 8);
            *(Q16*)(&sB[n * 32 + kb])     = b0;
            *(Q16*)(&sB[n * 32 + kb + 8]) = b1;
#endif
            if (k0 + 32 < K) __builtin_prefetch(src + 32, 0, 1);
        }
        __syncthreads();

        FragA af;
        const u16* ap = &sA[(wave * 16 + ln) * 32 + hl * 16];
        af.q[0] = *(const Q16*)ap;
        af.q[1] = *(const Q16*)(ap + 8);

        #pragma unroll
        for (int g = 0; g < 8; ++g) {
            v16bf bf = *(const v16bf*)(&sB[(g * 16 + ln) * 32 + hl * 16]);
            acc[g] = wmma_bf16(af.v, bf, acc[g]);
        }
        __syncthreads();
    }

    // epilogue: D layout (row = r + 8*hl, col = g*16 + ln)
    #pragma unroll
    for (int g = 0; g < 8; ++g) {
        int col = col0 + g * 16 + ln;
        float bv = bias[col];
        #pragma unroll
        for (int r = 0; r < 8; ++r) {
            long row = row0 + wave * 16 + hl * 8 + r;
            float v = (acc[g][r] + bv) * scale;
            if (RES_MODE == 1) {
                v += res[row * N + col];
            } else if (RES_MODE == 2) {
                int b = (int)(row >> 10), s = (int)(row & (SEQ - 1));
                v += res[((long)s * BATCH + b) * DMODEL + col];
            }
            if (RELU) v = fmaxf(v, 0.f);
            if (outF) {
                if (OUT_SEQ) {
                    int b = (int)(row >> 10), s = (int)(row & (SEQ - 1));
                    outF[((long)s * BATCH + b) * N + col] = v;
                } else {
                    outF[row * N + col] = v;
                }
            }
            if (outB) {
                if (OUTB_VT) {
                    int b = (int)(row >> 10), s = (int)(row & (SEQ - 1));
                    int hh = col >> 6, e = col & 63;
                    outB[(((long)b * NHEAD + hh) * HDIM + e) * SEQ + s] = f2bf(v);
                } else {
                    outB[row * N + col] = f2bf(v);
                }
            }
        }
    }
}

// ---------------------------------------------------------------------------
// Flash attention. Q (pre-scaled by 1/sqrt(HD)) and K: bf16 [B*S, H*HD].
// V: bf16 Vt[b][h][e][s].  O: bf16 [B*S, H*HD].
// 128 threads = 4 waves x 16 Q rows; K/V streamed in 32-key tiles.
// ---------------------------------------------------------------------------
__global__ __launch_bounds__(128) void flash_attn_kernel(
    const u16* __restrict__ Q, const u16* __restrict__ Km,
    const u16* __restrict__ Vt, u16* __restrict__ O)
{
    __shared__ __align__(32) u16 sK[32 * 64];       // [key][dim]
    __shared__ __align__(32) u16 sV[64 * 32];       // [col e][key]  (transposed)
    __shared__ __align__(32) u16 sP[4][16 * 32];    // per-wave P staging

    const int tid  = threadIdx.x;
    const int wave = tid >> 5;
    const int lane = tid & 31;
    const int hl   = lane >> 4;
    const int ln   = lane & 15;
    const int bh   = blockIdx.y;
    const int b    = bh >> 3;
    const int h    = bh & 7;
    const int q0   = blockIdx.x * 64 + wave * 16;
    const long baseRow = (long)b * SEQ;
    const int NC = NHEAD * HDIM;   // 512
    const u16* vbase = Vt + ((long)b * NHEAD + h) * HDIM * SEQ;

    // Q fragments (HD=64 -> two K=32 chunks), resident for whole kernel
    FragA qf[2];
    {
        const u16* qp = Q + (baseRow + q0 + ln) * NC + h * HDIM;
        #pragma unroll
        for (int c = 0; c < 2; ++c) {
            const u16* p = qp + c * 32 + hl * 16;
            qf[c].q[0] = *(const Q16*)p;
            qf[c].q[1] = *(const Q16*)(p + 8);
        }
    }

    v8f o[4] = {};
    float mrow[8], lrow[8];
    #pragma unroll
    for (int r = 0; r < 8; ++r) { mrow[r] = -1e30f; lrow[r] = 0.f; }

    for (int j = 0; j < SEQ; j += 32) {
        // ---- stage K tile (32 keys x 64 dims) and V tile (64 e x 32 keys) ----
        {
            int key = tid >> 2, db = (tid & 3) * 16;               // K mapping
            const u16* ks = Km + (baseRow + j + key) * NC + h * HDIM + db;
            int e = tid >> 1, kb = (tid & 1) * 16;                 // V mapping
            const u16* vs = vbase + (long)e * SEQ + j + kb;
#if USE_ASYNC_LDS
            __builtin_amdgcn_global_load_async_to_lds_b128(
                ASYNC_G(ks),     ASYNC_L(&sK[key * 64 + db]),     0, 0);
            __builtin_amdgcn_global_load_async_to_lds_b128(
                ASYNC_G(ks + 8), ASYNC_L(&sK[key * 64 + db + 8]), 0, 0);
            __builtin_amdgcn_global_load_async_to_lds_b128(
                ASYNC_G(vs),     ASYNC_L(&sV[e * 32 + kb]),       0, 0);
            __builtin_amdgcn_global_load_async_to_lds_b128(
                ASYNC_G(vs + 8), ASYNC_L(&sV[e * 32 + kb + 8]),   0, 0);
            __builtin_amdgcn_s_wait_asynccnt(0);
#else
            Q16 k0v = *(const Q16*)ks; Q16 k1v = *(const Q16*)(ks + 8);
            Q16 v0v = *(const Q16*)vs; Q16 v1v = *(const Q16*)(vs + 8);
            *(Q16*)(&sK[key * 64 + db])     = k0v;
            *(Q16*)(&sK[key * 64 + db + 8]) = k1v;
            *(Q16*)(&sV[e * 32 + kb])       = v0v;
            *(Q16*)(&sV[e * 32 + kb + 8])   = v1v;
#endif
        }
        __syncthreads();

        // scores S[16 x 32] = Q[16 x 64] * K_tile^T : B frag (k=dim, n=key)
        v8f sc[2] = {};
        #pragma unroll
        for (int g = 0; g < 2; ++g) {
            #pragma unroll
            for (int c = 0; c < 2; ++c) {
                v16bf kf = *(const v16bf*)(&sK[(g * 16 + ln) * 64 + c * 32 + hl * 16]);
                sc[g] = wmma_bf16(qf[c].v, kf, sc[g]);
            }
        }

        // online softmax: row stats across 16 lanes of each half-wave
        float mnew[8];
        #pragma unroll
        for (int r = 0; r < 8; ++r) {
            float mx = fmaxf(sc[0][r], sc[1][r]);
            #pragma unroll
            for (int d = 1; d < 16; d <<= 1) mx = fmaxf(mx, __shfl_xor(mx, d, 16));
            mnew[r] = fmaxf(mrow[r], mx);
        }
        #pragma unroll
        for (int r = 0; r < 8; ++r) {
            float alpha = __expf(mrow[r] - mnew[r]);
            lrow[r] *= alpha;
            #pragma unroll
            for (int g = 0; g < 4; ++g) o[g][r] *= alpha;
            mrow[r] = mnew[r];
        }
        #pragma unroll
        for (int g = 0; g < 2; ++g) {
            #pragma unroll
            for (int r = 0; r < 8; ++r) sc[g][r] = __expf(sc[g][r] - mnew[r]);
        }
        #pragma unroll
        for (int r = 0; r < 8; ++r) {
            float sm = sc[0][r] + sc[1][r];
            #pragma unroll
            for (int d = 1; d < 16; d <<= 1) sm += __shfl_xor(sm, d, 16);
            lrow[r] += sm;
        }

        // P: D layout -> LDS -> A layout (wave-local; DS in-order per wave)
        #pragma unroll
        for (int g = 0; g < 2; ++g) {
            #pragma unroll
            for (int r = 0; r < 8; ++r)
                sP[wave][(hl * 8 + r) * 32 + g * 16 + ln] = f2bf(sc[g][r]);
        }
        FragA pf;
        {
            const u16* pp = &sP[wave][ln * 32 + hl * 16];
            pf.q[0] = *(const Q16*)pp;
            pf.q[1] = *(const Q16*)(pp + 8);
        }

        // O[16 x 64] += P[16 x 32] * V_tile : B frag (k=key, n=e) from sV[e][key]
        #pragma unroll
        for (int g = 0; g < 4; ++g) {
            v16bf vf = *(const v16bf*)(&sV[(g * 16 + ln) * 32 + hl * 16]);
            o[g] = wmma_bf16(pf.v, vf, o[g]);
        }
        __syncthreads();
    }

    // finalize: divide by l, write bf16
    #pragma unroll
    for (int g = 0; g < 4; ++g) {
        #pragma unroll
        for (int r = 0; r < 8; ++r) {
            long row = baseRow + q0 + hl * 8 + r;
            float val = o[g][r] / lrow[r];
            O[row * NC + h * HDIM + g * 16 + ln] = f2bf(val);
        }
    }
}

// ---------------------------------------------------------------------------
extern "C" void kernel_launch(void* const* d_in, const int* in_sizes, int n_in,
                              void* d_out, int out_size, void* d_ws, size_t ws_size,
                              hipStream_t stream) {
    (void)in_sizes; (void)n_in; (void)out_size; (void)ws_size;
    const float* seq = (const float*)d_in[0];
    const float* Wq  = (const float*)d_in[1];
    const float* bq  = (const float*)d_in[2];
    const float* Wk  = (const float*)d_in[3];
    const float* bk  = (const float*)d_in[4];
    const float* Wv  = (const float*)d_in[5];
    const float* bv  = (const float*)d_in[6];
    const float* Wo  = (const float*)d_in[7];
    const float* bo  = (const float*)d_in[8];
    const float* W1  = (const float*)d_in[9];
    const float* b1  = (const float*)d_in[10];
    const float* W2  = (const float*)d_in[11];
    const float* b2  = (const float*)d_in[12];
    const float* W3  = (const float*)d_in[13];
    const float* b3  = (const float*)d_in[14];
    float* out = (float*)d_out;

    char* ws = (char*)d_ws;
    size_t off = 0;
    auto alloc = [&](size_t bytes) -> void* {
        size_t o = (off + 255) & ~(size_t)255;
        off = o + bytes;
        return (void*)(ws + o);
    };

    u16* Xbf  = (u16*)alloc((size_t)MROWS * DMODEL * 2);
    u16* Wqb  = (u16*)alloc((size_t)DMODEL * DMODEL * 2);   // [512][512] transposed
    u16* Wkb  = (u16*)alloc((size_t)DMODEL * DMODEL * 2);
    u16* Wvb  = (u16*)alloc((size_t)DMODEL * DMODEL * 2);
    u16* Wob  = (u16*)alloc((size_t)DMODEL * DMODEL * 2);   // [N=512][K=512]
    u16* W1b  = (u16*)alloc((size_t)DMODEL * FFDIM * 2);    // [N=2048][K=512]
    u16* W2b  = (u16*)alloc((size_t)FFDIM * FFDIM * 2);     // [2048][2048]
    u16* W3b  = (u16*)alloc((size_t)FFDIM * DMODEL * 2);    // [N=512][K=2048]
    u16* Qb   = (u16*)alloc((size_t)MROWS * DMODEL * 2);
    u16* Kb   = (u16*)alloc((size_t)MROWS * DMODEL * 2);
    u16* Vtb  = (u16*)alloc((size_t)MROWS * DMODEL * 2);    // Vt[b][h][e][s]
    u16* Ob   = (u16*)alloc((size_t)MROWS * DMODEL * 2);
    float* attF = (float*)alloc((size_t)MROWS * DMODEL * 4);
    u16* attB = (u16*)alloc((size_t)MROWS * DMODEL * 2);
    u16* h1b  = (u16*)alloc((size_t)MROWS * FFDIM * 2);
    u16* h2b  = (u16*)alloc((size_t)MROWS * FFDIM * 2);

    const int T = 256;
    pack_seq_kernel<<<((long)SEQ * BATCH * DMODEL + T - 1) / T, T, 0, stream>>>(seq, Xbf);
    repack_qkv_kernel<<<(NHEAD * DMODEL * HDIM + T - 1) / T, T, 0, stream>>>(Wq, Wqb);
    repack_qkv_kernel<<<(NHEAD * DMODEL * HDIM + T - 1) / T, T, 0, stream>>>(Wk, Wkb);
    repack_qkv_kernel<<<(NHEAD * DMODEL * HDIM + T - 1) / T, T, 0, stream>>>(Wv, Wvb);

    const dim3 tb(32, 8);
    transpose_cvt_kernel<<<dim3(DMODEL / 32, DMODEL / 32), tb, 0, stream>>>(Wo, Wob, DMODEL, DMODEL);
    transpose_cvt_kernel<<<dim3(FFDIM / 32, DMODEL / 32), tb, 0, stream>>>(W1, W1b, DMODEL, FFDIM);
    transpose_cvt_kernel<<<dim3(FFDIM / 32, FFDIM / 32),  tb, 0, stream>>>(W2, W2b, FFDIM, FFDIM);
    transpose_cvt_kernel<<<dim3(DMODEL / 32, FFDIM / 32), tb, 0, stream>>>(W3, W3b, FFDIM, DMODEL);

    const dim3 blk(256);
    const dim3 gN512(DMODEL / 128, MROWS / 128);   // (4,128)
    const dim3 gN2048(FFDIM / 128, MROWS / 128);   // (16,128)

    // QKV projections (softmax scale folded into Q; V written transposed)
    gemm_bf16_kernel<0, false, false, false><<<gN512, blk, 0, stream>>>(
        Xbf, Wqb, bq, nullptr, nullptr, Qb, DMODEL, DMODEL, 0.125f);
    gemm_bf16_kernel<0, false, false, false><<<gN512, blk, 0, stream>>>(
        Xbf, Wkb, bk, nullptr, nullptr, Kb, DMODEL, DMODEL, 1.0f);
    gemm_bf16_kernel<0, false, false, true><<<gN512, blk, 0, stream>>>(
        Xbf, Wvb, bv, nullptr, nullptr, Vtb, DMODEL, DMODEL, 1.0f);

    // attention
    flash_attn_kernel<<<dim3(SEQ / 64, BATCH * NHEAD), dim3(128), 0, stream>>>(Qb, Kb, Vtb, Ob);

    // att = x + O @ Wo + bo (residual from sequence layout); f32 + bf16 outputs
    gemm_bf16_kernel<2, false, false, false><<<gN512, blk, 0, stream>>>(
        Ob, Wob, bo, seq, attF, attB, DMODEL, DMODEL, 1.0f);

    // FFN
    gemm_bf16_kernel<0, true, false, false><<<gN2048, blk, 0, stream>>>(
        attB, W1b, b1, nullptr, nullptr, h1b, FFDIM, DMODEL, 1.0f);
    gemm_bf16_kernel<0, true, false, false><<<gN2048, blk, 0, stream>>>(
        h1b, W2b, b2, nullptr, nullptr, h2b, FFDIM, FFDIM, 1.0f);

    // y = att + h2 @ W3 + b3, written in [S,B,D] layout
    gemm_bf16_kernel<1, false, true, false><<<gN512, blk, 0, stream>>>(
        h2b, W3b, b3, attF, out, nullptr, DMODEL, FFDIM, 1.0f);
}
